// DilatedAttention_44281112822237
// MI455X (gfx1250) — matrix operand order repaired and checked by reading, not compiled, not run on hardware
//
#include <hip/hip_runtime.h>
#include <hip/hip_fp16.h>

typedef _Float16 h16;
typedef __attribute__((ext_vector_type(16))) _Float16 v16h;
typedef __attribute__((ext_vector_type(8)))  _Float16 v8h;
typedef __attribute__((ext_vector_type(2)))  _Float16 v2h;
typedef __attribute__((ext_vector_type(8)))  float    v8f;

#define EMBED   1024
#define LSEQ    4096
#define NHEAD   16
#define HDIM    64
#define TOTFLAT 7168   // per-head compact sparse positions: 4096 + 2048 + 1024

static __device__ __forceinline__ v8f wmma16(v16h a, v16h b, v8f c) {
  return __builtin_amdgcn_wmma_f32_16x16x32_f16(false, a, false, b, (short)0, c,
                                                false, false);
}

// A-operand fragment (16x32 f16): lane L holds row M=L&15.
// halves j=0..15 -> K = b8 + j ; b8 = (L>=16)?16:0
static __device__ __forceinline__ v16h load_fragA(const h16* rowptr, int lane) {
  const int b8 = (lane & 16) ? 16 : 0;
  v8h lo = *(const v8h*)(rowptr + b8);
  v8h hi = *(const v8h*)(rowptr + b8 + 8);
  v16h f;
#pragma unroll
  for (int i = 0; i < 8; ++i) { f[i] = lo[i]; f[i + 8] = hi[i]; }
  return f;
}

// B-operand fragment (32x16 f16): lane L holds col N=L&15.
// halves j=0..15 -> K = j + ((L>=16)?16:0). rowptr points to row-major row of B^T.
static __device__ __forceinline__ v16h load_fragB(const h16* rowptr, int lane) {
  const int b16 = (lane & 16) ? 16 : 0;
  v8h lo = *(const v8h*)(rowptr + b16);
  v8h hi = *(const v8h*)(rowptr + b16 + 8);
  v16h f;
#pragma unroll
  for (int i = 0; i < 8; ++i) { f[i] = lo[i]; f[i + 8] = hi[i]; }
  return f;
}

__global__ void cvt_f32_f16(const float* __restrict__ in, h16* __restrict__ out,
                            int n) {
  int i = blockIdx.x * blockDim.x + threadIdx.x;
  if (i < n) out[i] = (h16)in[i];
}

// C = A(MxK) @ W^T + bias. A,W row-major f16. Wave computes 32x32 tile.
// mode 0: f16 row-major out (ld=N). mode 1: f16 transposed out[n*M+m]. mode 2: f32 row-major.
__global__ void gemm_xwt(const h16* __restrict__ A, const h16* __restrict__ W,
                         const float* __restrict__ bias, void* __restrict__ out,
                         const int M, const int N, const int K, const int mode) {
  const int lane = threadIdx.x & 31;
  const int wave = blockIdx.x * (blockDim.x >> 5) + (threadIdx.x >> 5);
  const int tn = N >> 5;
  const int mb = (wave / tn) << 5;
  const int nb = (wave % tn) << 5;
  const int lq = lane & 15;
  if (mb >= M) return;
  v8f c00 = {}, c01 = {}, c10 = {}, c11 = {};
  const h16* a0 = A + (size_t)(mb + lq) * K;
  const h16* a1 = a0 + (size_t)16 * K;
  const h16* w0 = W + (size_t)(nb + lq) * K;
  const h16* w1 = w0 + (size_t)16 * K;
  for (int kb = 0; kb < K; kb += 32) {
    v16h fa0 = load_fragA(a0 + kb, lane);
    v16h fa1 = load_fragA(a1 + kb, lane);
    v16h fb0 = load_fragB(w0 + kb, lane);
    v16h fb1 = load_fragB(w1 + kb, lane);
    c00 = wmma16(fa0, fb0, c00);
    c01 = wmma16(fa0, fb1, c01);
    c10 = wmma16(fa1, fb0, c10);
    c11 = wmma16(fa1, fb1, c11);
  }
  const int b8 = (lane & 16) ? 8 : 0;
  v8f cs[2][2] = {{c00, c01}, {c10, c11}};
#pragma unroll
  for (int tm2 = 0; tm2 < 2; ++tm2)
#pragma unroll
    for (int tn2 = 0; tn2 < 2; ++tn2) {
      const int n = nb + tn2 * 16 + lq;
      const float bv = bias[n];
#pragma unroll
      for (int rr = 0; rr < 8; ++rr) {
        const int mm = mb + tm2 * 16 + b8 + rr;
        const float v = cs[tm2][tn2][rr] + bv;
        if (mode == 0)      ((h16*)out)[(size_t)mm * N + n] = (h16)v;
        else if (mode == 1) ((h16*)out)[(size_t)n * M + mm] = (h16)v;
        else                ((float*)out)[(size_t)mm * N + n] = v;
      }
    }
}

// Gather dilated V^T rows into per-branch compact layout: Vtb[h*64+d][flat]
template <int DR>
__global__ void gather_vt(const h16* __restrict__ Vt, h16* __restrict__ Vtb) {
  constexpr int Lb = LSEQ / DR;
  constexpr int SL = 1024 * DR;
  int idx = blockIdx.x * blockDim.x + threadIdx.x;
  if (idx >= NHEAD * HDIM * Lb) return;
  const int flat = idx % Lb;
  const int hd = idx / Lb;
  const int h = hd >> 6;
  const int grp = (h * DR) >> 4;
  const int s = flat >> 10;
  const int i = flat & 1023;
  const int p = s * SL + i * DR + grp;
  Vtb[idx] = Vt[(size_t)hd * LSEQ + p];
}

// One wave = one 16-query tile of one (branch, segment, head) causal attention
// problem of length g=1024, D=64. Computes S^T = K Q^T so softmax + P^T repack
// need only intra-wave shuffles; accumulates O^T; writes compact O and LSE.
template <int DR>
__global__ void attn_branch(const h16* __restrict__ Qm, const h16* __restrict__ Km,
                            const h16* __restrict__ Vt, float* __restrict__ o_ws,
                            float* __restrict__ lse_ws, const int boff) {
  constexpr int SL = 1024 * DR;
  constexpr int Lb = LSEQ / DR;
  const int lane = threadIdx.x & 31;
  const int wave = blockIdx.x * (blockDim.x >> 5) + (threadIdx.x >> 5);
  const int qt = wave & 63;
  const int h  = (wave >> 6) & 15;
  const int s  = wave >> 10;
  const int qb = qt << 4;
  const int grp = (h * DR) >> 4;
  const int lq  = lane & 15;
  const int b8  = (lane & 16) ? 8 : 0;
  const int q_row = qb + lq;

  const h16* qptr = Qm + ((size_t)s * SL + (size_t)q_row * DR + grp) * EMBED + h * HDIM;
  v16h fq0 = load_fragB(qptr, lane);        // Q^T as B operand over d=0..31
  v16h fq1 = load_fragB(qptr + 32, lane);   // d=32..63

  const h16* vrow = Vt + (size_t)(h * HDIM + lq) * Lb + (size_t)s * 1024;

  v8f o0 = {}, o1 = {}, o2 = {}, o3 = {};
  float m = -1e30f, l = 0.f;
  const float scale = 0.125f;               // 1/sqrt(64)
  const int nsteps = (qb + 47) >> 5;        // causal: keys 0 .. qb+15

  for (int ks = 0; ks < nsteps; ++ks) {
    const int kb = ks << 5;
    const h16* kp0 = Km + ((size_t)s * SL + (size_t)(kb + lq) * DR + grp) * EMBED + h * HDIM;
    const h16* kp1 = kp0 + (size_t)16 * DR * EMBED;
    v8f s0 = {}, s1 = {};
    s0 = wmma16(load_fragA(kp0, lane),      fq0, s0);
    s0 = wmma16(load_fragA(kp0 + 32, lane), fq1, s0);
    s1 = wmma16(load_fragA(kp1, lane),      fq0, s1);
    s1 = wmma16(load_fragA(kp1 + 32, lane), fq1, s1);

    float mloc = -1e30f;
#pragma unroll
    for (int rr = 0; rr < 8; ++rr) {
      const int k0 = kb + b8 + rr;        // S^T C-layout: reg rr -> key row
      const int k1 = k0 + 16;
      const float v0 = (k0 <= q_row) ? s0[rr] * scale : -1e30f;
      const float v1 = (k1 <= q_row) ? s1[rr] * scale : -1e30f;
      s0[rr] = v0; s1[rr] = v1;
      mloc = fmaxf(mloc, fmaxf(v0, v1));
    }
    mloc = fmaxf(mloc, __shfl_xor(mloc, 16, 32));
    const float mnew = fmaxf(m, mloc);

    v16h fp;                              // P^T as B operand (32 keys x 16 queries)
    float lloc = 0.f;
#pragma unroll
    for (int rr = 0; rr < 8; ++rr) {
      const float p0 = __expf(s0[rr] - mnew);
      const float p1 = __expf(s1[rr] - mnew);
      lloc += p0 + p1;
      v2h own; own[0] = (h16)p0; own[1] = (h16)p1;
      const unsigned pk = __builtin_bit_cast(unsigned, own);
      const unsigned po = (unsigned)__shfl_xor((int)pk, 16, 32);
      const v2h oth = __builtin_bit_cast(v2h, po);
      fp[rr]     = (lane & 16) ? oth[1] : own[0];
      fp[rr + 8] = (lane & 16) ? own[1] : oth[0];
    }
    lloc += __shfl_xor(lloc, 16, 32);
    const float fac = __expf(m - mnew);
    l = l * fac + lloc;
    m = mnew;
#pragma unroll
    for (int rr = 0; rr < 8; ++rr) {
      o0[rr] *= fac; o1[rr] *= fac; o2[rr] *= fac; o3[rr] *= fac;
    }
    o0 = wmma16(load_fragA(vrow + kb, lane),                     fp, o0);
    o1 = wmma16(load_fragA(vrow + (size_t)16 * Lb + kb, lane),   fp, o1);
    o2 = wmma16(load_fragA(vrow + (size_t)32 * Lb + kb, lane),   fp, o2);
    o3 = wmma16(load_fragA(vrow + (size_t)48 * Lb + kb, lane),   fp, o3);
  }

  const float linv = 1.f / l;
  const int flat = s * 1024 + qb + lq;
  float* orow = o_ws + ((size_t)h * TOTFLAT + boff + flat) * HDIM + b8;
#pragma unroll
  for (int rr = 0; rr < 8; ++rr) {
    orow[rr]      = o0[rr] * linv;
    orow[16 + rr] = o1[rr] * linv;
    orow[32 + rr] = o2[rr] * linv;
    orow[48 + rr] = o3[rr] * linv;
  }
  if (lane < 16)
    lse_ws[(size_t)h * TOTFLAT + boff + flat] = m + __logf(l);
}

// Softmax-merge the 3 branches per (l,h,d); coverage computed analytically.
__global__ void merge_branches(const float* __restrict__ o_ws,
                               const float* __restrict__ lse_ws,
                               h16* __restrict__ merged) {
  int idx = blockIdx.x * blockDim.x + threadIdx.x;
  if (idx >= LSEQ * EMBED) return;
  const int d = idx & 63;
  const int h = (idx >> 6) & 15;
  const int l = idx >> 10;
  float lse[3]; int fl[3]; bool cov[3];
  float mx = -1e30f;
#pragma unroll
  for (int b = 0; b < 3; ++b) {
    const int dr = 1 << b;
    const int sl = 1024 << b;
    const int bo = (b == 0) ? 0 : (b == 1 ? 4096 : 6144);
    const int grp = (h * dr) >> 4;
    const int ls = l & (sl - 1);
    const bool c = ((ls & (dr - 1)) == grp);
    const int i = (ls - grp) >> b;
    const int flat = (l >> (10 + b)) * 1024 + i;
    cov[b] = c;
    fl[b] = bo + flat;
    const float e = c ? lse_ws[(size_t)h * TOTFLAT + fl[b]] : -1e8f;
    lse[b] = e;
    mx = fmaxf(mx, e);
  }
  float wsum = 0.f, acc = 0.f;
#pragma unroll
  for (int b = 0; b < 3; ++b) {
    const float w = __expf(lse[b] - mx);
    wsum += w;
    if (cov[b]) acc += w * o_ws[((size_t)h * TOTFLAT + fl[b]) * HDIM + d];
  }
  merged[(size_t)l * EMBED + h * HDIM + d] = (h16)(acc / wsum);
}

extern "C" void kernel_launch(void* const* d_in, const int* in_sizes, int n_in,
                              void* d_out, int out_size, void* d_ws, size_t ws_size,
                              hipStream_t stream) {
  const float* query = (const float*)d_in[0];
  const float* key   = (const float*)d_in[1];
  const float* value = (const float*)d_in[2];
  const float* Wq = (const float*)d_in[3];
  const float* bq = (const float*)d_in[4];
  const float* Wk = (const float*)d_in[5];
  const float* bk = (const float*)d_in[6];
  const float* Wv = (const float*)d_in[7];
  const float* bv = (const float*)d_in[8];
  const float* Wo = (const float*)d_in[9];
  const float* bo = (const float*)d_in[10];

  char* ws = (char*)d_ws;
  size_t off = 0;
  auto alloc = [&](size_t bytes) -> char* {
    char* p = ws + off;
    off += (bytes + 255) & ~(size_t)255;
    return p;
  };
  const size_t NX = (size_t)LSEQ * EMBED;     // 4M activation elements
  const size_t NW = (size_t)EMBED * EMBED;    // 1M weight elements
  h16* qh   = (h16*)alloc(NX * 2);
  h16* kh   = (h16*)alloc(NX * 2);
  h16* vh   = (h16*)alloc(NX * 2);
  h16* wqh  = (h16*)alloc(NW * 2);
  h16* wkh  = (h16*)alloc(NW * 2);
  h16* wvh  = (h16*)alloc(NW * 2);
  h16* woh  = (h16*)alloc(NW * 2);
  h16* Qm   = (h16*)alloc(NX * 2);
  h16* Km   = (h16*)alloc(NX * 2);
  h16* Vt   = (h16*)alloc(NX * 2);                            // [h*64+d][4096]
  h16* Vtb1 = (h16*)alloc((size_t)NHEAD * HDIM * 2048 * 2);   // branch dr=2
  h16* Vtb2 = (h16*)alloc((size_t)NHEAD * HDIM * 1024 * 2);   // branch dr=4
  float* o_f   = (float*)alloc((size_t)NHEAD * TOTFLAT * HDIM * 4);
  float* lse_f = (float*)alloc((size_t)NHEAD * TOTFLAT * 4);
  h16* merged  = (h16*)alloc(NX * 2);

  // 1) f32 -> f16 conversions
  cvt_f32_f16<<<(int)((NX + 255) / 256), 256, 0, stream>>>(query, qh, (int)NX);
  cvt_f32_f16<<<(int)((NX + 255) / 256), 256, 0, stream>>>(key,   kh, (int)NX);
  cvt_f32_f16<<<(int)((NX + 255) / 256), 256, 0, stream>>>(value, vh, (int)NX);
  cvt_f32_f16<<<(int)((NW + 255) / 256), 256, 0, stream>>>(Wq, wqh, (int)NW);
  cvt_f32_f16<<<(int)((NW + 255) / 256), 256, 0, stream>>>(Wk, wkh, (int)NW);
  cvt_f32_f16<<<(int)((NW + 255) / 256), 256, 0, stream>>>(Wv, wvh, (int)NW);
  cvt_f32_f16<<<(int)((NW + 255) / 256), 256, 0, stream>>>(Wo, woh, (int)NW);

  // 2) QKV projections: (4096/32)*(1024/32)=4096 waves, 8 waves/block -> 512 blocks
  const int gemm_blocks = (LSEQ / 32) * (EMBED / 32) / 8;
  gemm_xwt<<<gemm_blocks, 256, 0, stream>>>(qh, wqh, bq, Qm, LSEQ, EMBED, EMBED, 0);
  gemm_xwt<<<gemm_blocks, 256, 0, stream>>>(kh, wkh, bk, Km, LSEQ, EMBED, EMBED, 0);
  gemm_xwt<<<gemm_blocks, 256, 0, stream>>>(vh, wvh, bv, Vt, LSEQ, EMBED, EMBED, 1);

  // 3) Per-branch V^T gathers (dr=1 uses Vt directly)
  gather_vt<2><<<(NHEAD * HDIM * 2048) / 256, 256, 0, stream>>>(Vt, Vtb1);
  gather_vt<4><<<(NHEAD * HDIM * 1024) / 256, 256, 0, stream>>>(Vt, Vtb2);

  // 4) Attention branches: waves = nseg*16*64, 4 waves/block (128 threads)
  attn_branch<1><<<(4 * 16 * 64) / 4, 128, 0, stream>>>(Qm, Km, Vt,   o_f, lse_f, 0);
  attn_branch<2><<<(2 * 16 * 64) / 4, 128, 0, stream>>>(Qm, Km, Vtb1, o_f, lse_f, 4096);
  attn_branch<4><<<(1 * 16 * 64) / 4, 128, 0, stream>>>(Qm, Km, Vtb2, o_f, lse_f, 6144);

  // 5) Softmax merge of branches -> f16 merged activations
  merge_branches<<<(int)(NX / 256), 256, 0, stream>>>(o_f, lse_f, merged);

  // 6) Output projection -> f32 d_out
  gemm_xwt<<<gemm_blocks, 256, 0, stream>>>(merged, woh, bo, (float*)d_out,
                                            LSEQ, EMBED, EMBED, 2);
}
